// _PointNet2SAModuleBase_66056597012940
// MI455X (gfx1250) — matrix operand order repaired and checked by reading, not compiled, not run on hardware
//
#include <hip/hip_runtime.h>
#include <hip/hip_bf16.h>

// ---------------------------------------------------------------------------
// PointNet++ Set Abstraction (FPS -> kNN -> grouped MLP -> maxpool) on gfx1250
// B=8, N=16384, C=64, NPOINT=1024, NSAMPLE=32, MLP [67,64,64,128]
// ---------------------------------------------------------------------------

#define PN_B 8
#define PN_N 16384
#define PN_C 64
#define PN_P 1024
#define PN_K 32

typedef __attribute__((ext_vector_type(16))) _Float16 v16h;
typedef __attribute__((ext_vector_type(8)))  float    v8f;

// ===========================================================================
// Kernel 1: farthest point sampling. One 1024-thread block per batch.
// Points + running min-dist live in registers (16 per thread).
// ===========================================================================
__global__ __launch_bounds__(1024) void fps_kernel(const float* __restrict__ xyz,
                                                   int* __restrict__ center_idx,
                                                   float* __restrict__ new_xyz) {
  const int b = blockIdx.x;
  const int tid = threadIdx.x;
  const float* xb = xyz + (size_t)b * PN_N * 3;

  float px[16], py[16], pz[16], dd[16];
#pragma unroll
  for (int i = 0; i < 16; ++i) {
    const int n = tid + (i << 10);
    px[i] = xb[n * 3 + 0];
    py[i] = xb[n * 3 + 1];
    pz[i] = xb[n * 3 + 2];
    dd[i] = 1.0e10f;
  }

  __shared__ float s_c[3];
  __shared__ float s_wv[32];
  __shared__ int   s_wi[32];

  int*   ci = center_idx + b * PN_P;
  float* nx = new_xyz + (size_t)b * PN_P * 3;
  if (tid == 0) {
    ci[0] = 0;
    s_c[0] = xb[0]; s_c[1] = xb[1]; s_c[2] = xb[2];
    nx[0] = xb[0];  nx[1] = xb[1];  nx[2] = xb[2];
  }
  __syncthreads();
  float cx = s_c[0], cy = s_c[1], cz = s_c[2];

  const int wid = tid >> 5, lane = tid & 31;

  for (int it = 1; it < PN_P; ++it) {
    // update running min distance, track local argmax
    float bv = -1.0f;
    int   bi = 0;
#pragma unroll
    for (int i = 0; i < 16; ++i) {
      const int n = tid + (i << 10);
      const float dx = px[i] - cx, dy = py[i] - cy, dz = pz[i] - cz;
      float d = dx * dx + dy * dy + dz * dz;
      d = fminf(dd[i], d);
      dd[i] = d;
      if (d > bv) { bv = d; bi = n; }
    }
    // wave32 argmax reduce (prefer smaller index on ties, like argmax)
#pragma unroll
    for (int off = 16; off > 0; off >>= 1) {
      const float ov = __shfl_xor(bv, off);
      const int   oi = __shfl_xor(bi, off);
      if (ov > bv || (ov == bv && oi < bi)) { bv = ov; bi = oi; }
    }
    if (lane == 0) { s_wv[wid] = bv; s_wi[wid] = bi; }
    __syncthreads();
    if (wid == 0) {
      bv = s_wv[lane]; bi = s_wi[lane];
#pragma unroll
      for (int off = 16; off > 0; off >>= 1) {
        const float ov = __shfl_xor(bv, off);
        const int   oi = __shfl_xor(bi, off);
        if (ov > bv || (ov == bv && oi < bi)) { bv = ov; bi = oi; }
      }
      if (lane == 0) {
        ci[it] = bi;
        const float wx = xb[bi * 3 + 0], wy = xb[bi * 3 + 1], wz = xb[bi * 3 + 2];
        nx[it * 3 + 0] = wx; nx[it * 3 + 1] = wy; nx[it * 3 + 2] = wz;
        s_c[0] = wx; s_c[1] = wy; s_c[2] = wz;
      }
    }
    __syncthreads();
    cx = s_c[0]; cy = s_c[1]; cz = s_c[2];
  }
}

// ===========================================================================
// Kernel 2: 32-NN per center. One 256-thread block per center.
// Each thread owns 64 candidate points; iterative block argmin where only
// the winning thread rescans (distances recomputed from L2-resident xyz).
// ===========================================================================
__global__ __launch_bounds__(256) void knn_kernel(const float* __restrict__ xyz,
                                                  const float* __restrict__ new_xyz,
                                                  int* __restrict__ knn_idx) {
  __shared__ float s_wv[8];
  __shared__ int   s_wi[8];
  __shared__ int   s_winner;

  const int tid = threadIdx.x;
  const int b = blockIdx.x >> 10;
  const int p = blockIdx.x & 1023;
  const float* xb = xyz + (size_t)b * PN_N * 3;
  const float* cp = new_xyz + (size_t)(b * PN_P + p) * 3;
  const float cx = cp[0], cy = cp[1], cz = cp[2];

  unsigned long long excl = 0ULL;
  float mv = 3.0e38f;
  int   mi = tid;
  for (int k = 0; k < 64; ++k) {
    const int n = tid + (k << 8);
    const float dx = xb[n * 3 + 0] - cx, dy = xb[n * 3 + 1] - cy, dz = xb[n * 3 + 2] - cz;
    const float d = dx * dx + dy * dy + dz * dz;
    if (d < mv || (d == mv && n < mi)) { mv = d; mi = n; }
  }

  const int wid = tid >> 5, lane = tid & 31;
  int* ob = knn_idx + (size_t)(b * PN_P + p) * PN_K;

  for (int s = 0; s < PN_K; ++s) {
    float rv = mv;
    int   ri = mi;
#pragma unroll
    for (int off = 16; off > 0; off >>= 1) {
      const float ov = __shfl_xor(rv, off);
      const int   oi = __shfl_xor(ri, off);
      if (ov < rv || (ov == rv && oi < ri)) { rv = ov; ri = oi; }
    }
    if (lane == 0) { s_wv[wid] = rv; s_wi[wid] = ri; }
    __syncthreads();
    if (tid == 0) {
      float bvv = s_wv[0];
      int   bii = s_wi[0];
      for (int w = 1; w < 8; ++w) {
        if (s_wv[w] < bvv || (s_wv[w] == bvv && s_wi[w] < bii)) { bvv = s_wv[w]; bii = s_wi[w]; }
      }
      ob[s] = bii;
      s_winner = bii;
    }
    __syncthreads();
    const int win = s_winner;
    if ((win & 255) == tid) {
      excl |= 1ULL << (win >> 8);
      mv = 3.0e38f;
      mi = tid;
      for (int k = 0; k < 64; ++k) {
        if ((excl >> k) & 1ULL) continue;
        const int n = tid + (k << 8);
        const float dx = xb[n * 3 + 0] - cx, dy = xb[n * 3 + 1] - cy, dz = xb[n * 3 + 2] - cz;
        const float d = dx * dx + dy * dy + dz * dz;
        if (d < mv || (d == mv && n < mi)) { mv = d; mi = n; }
      }
    }
  }
}

// ===========================================================================
// Kernel 3: gather + 3-layer shared MLP (WMMA f16, f32 accum) + neighbor max.
// 128 threads = 4 waves; one (b,p) group per wave (M=32 neighbors).
// Weights pre-swizzled into a contiguous-k B-fragment layout at preload so each
// lane's B-frag is a single contiguous 32-byte LDS read.
// ===========================================================================
__device__ __forceinline__ v8f wmma_f16(v16h a, v16h b, v8f c) {
  // (neg_a, A, neg_b, B, c_mod, C, reuse_a, reuse_b)
  return __builtin_amdgcn_wmma_f32_16x16x32_f16(false, a, false, b, (short)0, c, false, false);
}

// B-tile swizzle: element j of lane ln in tile (kt,nt) is
//   k = kt*32 + (ln>=16 ? 16 : 0) + j ,  n = nt*16 + (ln&15)
__device__ __forceinline__ void load_w_swz(_Float16* dst, const float* __restrict__ w,
                                           int KT, int NT, int cin, int tid) {
  const int total = KT * NT * 512;
  for (int e = tid; e < total; e += 128) {
    const int j  = e & 15;
    const int ln = (e >> 4) & 31;
    const int nt = (e >> 9) % NT;
    const int kt = e / (NT << 9);
    const int k = kt * 32 + ((ln >> 4) << 4) + j;
    const int n = nt * 16 + (ln & 15);
    const float v = (k < cin) ? w[n * cin + k] : 0.0f;
    dst[e] = (_Float16)v;
  }
}

// A-frag (16x32 f16): lane<16 -> K = kb..kb+15, lane>=16 -> K = kb+16..kb+31 (matches B swizzle)
__device__ __forceinline__ v16h load_a_frag(const _Float16* act, int mt, int kt, int lane) {
  const int m  = mt * 16 + (lane & 15);
  const int kb = kt * 32 + ((lane >> 4) << 4);
  const _Float16* r = act + m * 96 + kb;
  v16h a;
#pragma unroll
  for (int j = 0; j < 16; ++j) { a[j] = r[j]; }
  return a;
}

__device__ __forceinline__ v16h load_b_frag(const _Float16* wbuf, int NT, int kt, int nt, int lane) {
  const _Float16* r = wbuf + (((kt * NT + nt) * 32 + lane) << 4);
  return *(const v16h*)r;
}

__global__ __launch_bounds__(128) void sa_mlp_kernel(
    const float* __restrict__ xyz, const float* __restrict__ features,
    const float* __restrict__ w0, const float* __restrict__ s0v, const float* __restrict__ b0v,
    const float* __restrict__ w1, const float* __restrict__ s1v, const float* __restrict__ b1v,
    const float* __restrict__ w2, const float* __restrict__ s2v, const float* __restrict__ b2v,
    const float* __restrict__ new_xyz, const int* __restrict__ knn_idx,
    float* __restrict__ out) {
  __shared__ __attribute__((aligned(32))) _Float16 s_w0[3 * 4 * 512];   // 96x64
  __shared__ __attribute__((aligned(32))) _Float16 s_w1[2 * 4 * 512];   // 64x64
  __shared__ __attribute__((aligned(32))) _Float16 s_w2[2 * 8 * 512];   // 64x128
  __shared__ __attribute__((aligned(32))) _Float16 s_act[4][32 * 96];   // per-wave activations
  __shared__ float s_sb[512];  // s0 b0 s1 b1 | s2(128) b2(128)

  const int tid  = threadIdx.x;
  const int wave = tid >> 5;
  const int lane = tid & 31;

  load_w_swz(s_w0, w0, 3, 4, 67, tid);
  load_w_swz(s_w1, w1, 2, 4, 64, tid);
  load_w_swz(s_w2, w2, 2, 8, 64, tid);
  if (tid < 64) {
    s_sb[tid]       = s0v[tid];
    s_sb[64 + tid]  = b0v[tid];
    s_sb[128 + tid] = s1v[tid];
    s_sb[192 + tid] = b1v[tid];
  }
  if (tid < 128) {
    s_sb[256 + tid] = s2v[tid];
    s_sb[384 + tid] = b2v[tid];
  }
  __syncthreads();

  const int gid = blockIdx.x * 4 + wave;   // group = (b,p)
  const int b = gid >> 10;
  const int p = gid & 1023;
  _Float16* act = s_act[wave];             // wave-private: no barriers needed below

  // ---- gather: lane = neighbor, row = [rel_xyz(3) | feats(64) | pad(29)] ----
  {
    const int nidx = knn_idx[(size_t)gid * PN_K + lane];
    const float* xb = xyz + (size_t)b * PN_N * 3;
    const float* cp = new_xyz + (size_t)(b * PN_P + p) * 3;
    _Float16* row = act + lane * 96;
    row[0] = (_Float16)(xb[nidx * 3 + 0] - cp[0]);
    row[1] = (_Float16)(xb[nidx * 3 + 1] - cp[1]);
    row[2] = (_Float16)(xb[nidx * 3 + 2] - cp[2]);
    const float* fb = features + (size_t)b * PN_C * PN_N + nidx;
#pragma unroll 8
    for (int c = 0; c < PN_C; ++c) row[3 + c] = (_Float16)fb[(size_t)c * PN_N];
#pragma unroll
    for (int c = 67; c < 96; ++c) row[c] = (_Float16)0.0f;
  }

  // ---- layer 1: (32x96) @ (96x64) -> 32x64, BN+ReLU, back into act ----
  {
    v8f acc[2][4];
#pragma unroll
    for (int mt = 0; mt < 2; ++mt)
#pragma unroll
      for (int nt = 0; nt < 4; ++nt) acc[mt][nt] = (v8f){0, 0, 0, 0, 0, 0, 0, 0};
#pragma unroll
    for (int kt = 0; kt < 3; ++kt) {
      const v16h a0 = load_a_frag(act, 0, kt, lane);
      const v16h a1 = load_a_frag(act, 1, kt, lane);
#pragma unroll
      for (int nt = 0; nt < 4; ++nt) {
        const v16h bf = load_b_frag(s_w0, 4, kt, nt, lane);
        acc[0][nt] = wmma_f16(a0, bf, acc[0][nt]);
        acc[1][nt] = wmma_f16(a1, bf, acc[1][nt]);
      }
    }
#pragma unroll
    for (int mt = 0; mt < 2; ++mt) {
      const int m0 = mt * 16 + ((lane >> 4) << 3);
#pragma unroll
      for (int nt = 0; nt < 4; ++nt) {
        const int n = nt * 16 + (lane & 15);
        const float sc = s_sb[n], bi = s_sb[64 + n];
#pragma unroll
        for (int r = 0; r < 8; ++r) {
          float v = acc[mt][nt][r] * sc + bi;
          v = v > 0.0f ? v : 0.0f;
          act[(m0 + r) * 96 + n] = (_Float16)v;
        }
      }
    }
  }

  // ---- layer 2: (32x64) @ (64x64) -> 32x64, BN+ReLU, back into act ----
  {
    v8f acc[2][4];
#pragma unroll
    for (int mt = 0; mt < 2; ++mt)
#pragma unroll
      for (int nt = 0; nt < 4; ++nt) acc[mt][nt] = (v8f){0, 0, 0, 0, 0, 0, 0, 0};
#pragma unroll
    for (int kt = 0; kt < 2; ++kt) {
      const v16h a0 = load_a_frag(act, 0, kt, lane);
      const v16h a1 = load_a_frag(act, 1, kt, lane);
#pragma unroll
      for (int nt = 0; nt < 4; ++nt) {
        const v16h bf = load_b_frag(s_w1, 4, kt, nt, lane);
        acc[0][nt] = wmma_f16(a0, bf, acc[0][nt]);
        acc[1][nt] = wmma_f16(a1, bf, acc[1][nt]);
      }
    }
#pragma unroll
    for (int mt = 0; mt < 2; ++mt) {
      const int m0 = mt * 16 + ((lane >> 4) << 3);
#pragma unroll
      for (int nt = 0; nt < 4; ++nt) {
        const int n = nt * 16 + (lane & 15);
        const float sc = s_sb[128 + n], bi = s_sb[192 + n];
#pragma unroll
        for (int r = 0; r < 8; ++r) {
          float v = acc[mt][nt][r] * sc + bi;
          v = v > 0.0f ? v : 0.0f;
          act[(m0 + r) * 96 + n] = (_Float16)v;
        }
      }
    }
  }

  // ---- layer 3: (32x64) @ (64x128) -> 32x128, BN+ReLU, max over 32 rows ----
#pragma unroll
  for (int nh = 0; nh < 2; ++nh) {   // two halves of 64 output channels
    v8f acc[2][4];
#pragma unroll
    for (int mt = 0; mt < 2; ++mt)
#pragma unroll
      for (int nt = 0; nt < 4; ++nt) acc[mt][nt] = (v8f){0, 0, 0, 0, 0, 0, 0, 0};
#pragma unroll
    for (int kt = 0; kt < 2; ++kt) {
      const v16h a0 = load_a_frag(act, 0, kt, lane);
      const v16h a1 = load_a_frag(act, 1, kt, lane);
#pragma unroll
      for (int nt = 0; nt < 4; ++nt) {
        const v16h bf = load_b_frag(s_w2, 8, kt, nh * 4 + nt, lane);
        acc[0][nt] = wmma_f16(a0, bf, acc[0][nt]);
        acc[1][nt] = wmma_f16(a1, bf, acc[1][nt]);
      }
    }
#pragma unroll
    for (int nt = 0; nt < 4; ++nt) {
      const int n = (nh * 4 + nt) * 16 + (lane & 15);
      const float sc = s_sb[256 + n], bi = s_sb[384 + n];
      float mx = 0.0f;  // ReLU outputs are >= 0, so 0 is the exact identity
#pragma unroll
      for (int mt = 0; mt < 2; ++mt)
#pragma unroll
        for (int r = 0; r < 8; ++r) {
          float v = acc[mt][nt][r] * sc + bi;
          v = v > 0.0f ? v : 0.0f;
          mx = v > mx ? v : mx;
        }
      // merge rows 0-7 (lanes<16) with rows 8-15 (lanes>=16) of each tile
      mx = fmaxf(mx, __shfl_xor(mx, 16));
      if (lane < 16) out[((size_t)b * 128 + n) * PN_P + p] = mx;
    }
  }
}

// ===========================================================================
extern "C" void kernel_launch(void* const* d_in, const int* in_sizes, int n_in,
                              void* d_out, int out_size, void* d_ws, size_t ws_size,
                              hipStream_t stream) {
  const float* xyz      = (const float*)d_in[0];
  const float* features = (const float*)d_in[1];
  const float* w0 = (const float*)d_in[2];
  const float* s0 = (const float*)d_in[3];
  const float* b0 = (const float*)d_in[4];
  const float* w1 = (const float*)d_in[5];
  const float* s1 = (const float*)d_in[6];
  const float* b1 = (const float*)d_in[7];
  const float* w2 = (const float*)d_in[8];
  const float* s2 = (const float*)d_in[9];
  const float* b2 = (const float*)d_in[10];
  float* out = (float*)d_out;

  char* ws = (char*)d_ws;
  int*   center_idx = (int*)ws;                              // 8*1024 ints   (32 KB)
  float* new_xyz    = (float*)(ws + 32768);                  // 8*1024*3 f32  (96 KB)
  int*   knn_idx    = (int*)(ws + 32768 + 98304);            // 8*1024*32 int (1 MB)

  fps_kernel<<<PN_B, 1024, 0, stream>>>(xyz, center_idx, new_xyz);
  knn_kernel<<<PN_B * PN_P, 256, 0, stream>>>(xyz, new_xyz, knn_idx);
  sa_mlp_kernel<<<(PN_B * PN_P) / 4, 128, 0, stream>>>(
      xyz, features, w0, s0, b0, w1, s1, b1, w2, s2, b2, new_xyz, knn_idx, out);
}
